// Net_9783935500532
// MI455X (gfx1250) — compile-verified, matrix operand order checked by reading, not run on hardware
//
#include <hip/hip_runtime.h>
#include <math.h>

// ---------------- problem constants (from reference) ----------------
#define NNODES   20000
#define NEDGES   320000
#define INF_DIM  1280
#define NHEAD    4
#define DHEAD    128
#define HDDIM    512            // NHEAD * DHEAD
#define FC_OUT   64
#define NEG_SLOPE 0.2f

// ---------------- WMMA fragment types (gfx1250, wave32) ----------------
typedef __attribute__((ext_vector_type(16))) __bf16 v16bf;
typedef __attribute__((ext_vector_type(8)))  __bf16 v8bf;
typedef __attribute__((ext_vector_type(8)))  float  v8f;

union FragAB { v16bf v; v8bf h[2]; };

// ---------------- fp32 -> bf16 conversions ----------------
__global__ void k_cvt_bf16(const float* __restrict__ in, __bf16* __restrict__ out, int count) {
  int i = blockIdx.x * blockDim.x + threadIdx.x;
  if (i < count) out[i] = (__bf16)in[i];
}

// W is K x N row-major fp32; emit N x K row-major bf16 (transposed) so GEMM B
// fragments become two contiguous 16B loads per lane.
__global__ void k_cvt_transpose_bf16(const float* __restrict__ in, __bf16* __restrict__ out,
                                     int K, int N) {
  int i = blockIdx.x * blockDim.x + threadIdx.x;
  if (i >= K * N) return;
  int k = i / N, n = i - k * N;
  out[(size_t)n * K + k] = (__bf16)in[i];
}

// ---------------- bf16 WMMA GEMM: C[M,N] = A[M,K] * BT[N,K]^T ----------------
// One wave computes a 32x64 output block: 2 A fragments x 4 B fragments,
// each B fragment reused by two WMMAs -> 8 v_wmma per 32-deep K step.
// flags: bit0 = add bias[col], bit1 = ELU epilogue.
__global__ __launch_bounds__(128) void k_gemm_wmma(
    const __bf16* __restrict__ A,    // M x K row-major bf16
    const __bf16* __restrict__ BT,   // N x K row-major bf16 (i.e. B transposed)
    const float*  __restrict__ bias, // N (may be null if !(flags&1))
    float* __restrict__ C,           // M x N fp32
    int M, int N, int K, int flags)
{
  const int lane   = threadIdx.x & 31;
  const int wv     = blockIdx.x * (blockDim.x >> 5) + (threadIdx.x >> 5);
  const int ntiles = N >> 6;                 // 64-wide strips
  const int total  = (M >> 5) * ntiles;      // 32-row blocks
  if (wv >= total) return;
  const int rowt = wv / ntiles, colt = wv - rowt * ntiles;
  const int row0 = rowt << 5, col0 = colt << 6;

  // ISA 7.12.2 16-bit A layout: lanes 0-15 row M=lane, K {0..7,16..23};
  // lanes 16-31 same rows, K {8..15,24..31}.  (B mirrors with N in lane&15.)
  const int idx15 = lane & 15;
  const int kb    = (lane >> 4) << 3;        // 0 or 8

  const __bf16* ap0 = A  + (size_t)(row0 + idx15) * K + kb;        // rows 0..15
  const __bf16* ap1 = ap0 + (size_t)16 * K;                        // rows 16..31
  const __bf16* bp0 = BT + (size_t)(col0 + idx15) * K + kb;
  const size_t  bstep = (size_t)16 * K;      // next 16-column tile of BT

  v8f acc00 = {}, acc01 = {}, acc10 = {}, acc11 = {};
  v8f acc20 = {}, acc21 = {}, acc30 = {}, acc31 = {};

  for (int k = 0; k < K; k += 32) {
    FragAB a0, a1;
    a0.h[0] = *(const v8bf*)(ap0 + k);
    a0.h[1] = *(const v8bf*)(ap0 + k + 16);
    a1.h[0] = *(const v8bf*)(ap1 + k);
    a1.h[1] = *(const v8bf*)(ap1 + k + 16);
    const __bf16* bp = bp0 + k;
    FragAB b;
    b.h[0] = *(const v8bf*)(bp);
    b.h[1] = *(const v8bf*)(bp + 16);
    acc00 = __builtin_amdgcn_wmma_f32_16x16x32_bf16(false, a0.v, false, b.v, (short)0, acc00, false, false);
    acc01 = __builtin_amdgcn_wmma_f32_16x16x32_bf16(false, a1.v, false, b.v, (short)0, acc01, false, false);
    b.h[0] = *(const v8bf*)(bp + bstep);
    b.h[1] = *(const v8bf*)(bp + bstep + 16);
    acc10 = __builtin_amdgcn_wmma_f32_16x16x32_bf16(false, a0.v, false, b.v, (short)0, acc10, false, false);
    acc11 = __builtin_amdgcn_wmma_f32_16x16x32_bf16(false, a1.v, false, b.v, (short)0, acc11, false, false);
    b.h[0] = *(const v8bf*)(bp + 2 * bstep);
    b.h[1] = *(const v8bf*)(bp + 2 * bstep + 16);
    acc20 = __builtin_amdgcn_wmma_f32_16x16x32_bf16(false, a0.v, false, b.v, (short)0, acc20, false, false);
    acc21 = __builtin_amdgcn_wmma_f32_16x16x32_bf16(false, a1.v, false, b.v, (short)0, acc21, false, false);
    b.h[0] = *(const v8bf*)(bp + 3 * bstep);
    b.h[1] = *(const v8bf*)(bp + 3 * bstep + 16);
    acc30 = __builtin_amdgcn_wmma_f32_16x16x32_bf16(false, a0.v, false, b.v, (short)0, acc30, false, false);
    acc31 = __builtin_amdgcn_wmma_f32_16x16x32_bf16(false, a1.v, false, b.v, (short)0, acc31, false, false);
  }

  // C/D layout: N = lane&15; M = vgpr + 8*(lane>=16)
  const int crow0 = row0 + ((lane >> 4) << 3);
  const int ccol  = idx15;
#define STORE_TILE(ACC, T, RH)                                                 \
  {                                                                            \
    const int col = col0 + ((T) << 4) + ccol;                                  \
    const float bv = (flags & 1) ? bias[col] : 0.0f;                           \
    _Pragma("unroll") for (int r = 0; r < 8; ++r) {                            \
      float v = ACC[r] + bv;                                                   \
      if (flags & 2) v = v > 0.0f ? v : (__expf(v) - 1.0f);                    \
      C[(size_t)(crow0 + ((RH) << 4) + r) * N + col] = v;                      \
    }                                                                          \
  }
  STORE_TILE(acc00, 0, 0) STORE_TILE(acc01, 0, 1)
  STORE_TILE(acc10, 1, 0) STORE_TILE(acc11, 1, 1)
  STORE_TILE(acc20, 2, 0) STORE_TILE(acc21, 2, 1)
  STORE_TILE(acc30, 3, 0) STORE_TILE(acc31, 3, 1)
#undef STORE_TILE
}

// ---------------- attention scores: el/er = sum_d h[n,h,d]*a[h,d] ----------------
__global__ void k_scores(const float* __restrict__ h, const float* __restrict__ al,
                         const float* __restrict__ ar,
                         float* __restrict__ el, float* __restrict__ er) {
  int idx = blockIdx.x * blockDim.x + threadIdx.x;
  if (idx >= NNODES * NHEAD) return;
  int n = idx >> 2, hh = idx & 3;
  const float4* hp  = (const float4*)(h + (size_t)n * HDDIM + hh * DHEAD);
  const float4* alp = (const float4*)(al + hh * DHEAD);
  const float4* arp = (const float4*)(ar + hh * DHEAD);
  float sl = 0.f, sr = 0.f;
#pragma unroll 8
  for (int j = 0; j < DHEAD / 4; ++j) {
    float4 hv = hp[j], a = alp[j], b = arp[j];
    sl += hv.x * a.x + hv.y * a.y + hv.z * a.z + hv.w * a.w;
    sr += hv.x * b.x + hv.y * b.y + hv.z * b.z + hv.w * b.w;
  }
  el[idx] = sl;
  er[idx] = sr;
}

// ---------------- CSR row pointers from sorted dst (lower_bound) ----------------
__global__ void k_rowptr(const int* __restrict__ dst, int* __restrict__ rowptr) {
  int n = blockIdx.x * blockDim.x + threadIdx.x;
  if (n > NNODES) return;
  int lo = 0, hi = NEDGES;
  while (lo < hi) {
    int mid = (lo + hi) >> 1;
    if (dst[mid] < n) lo = mid + 1; else hi = mid;
  }
  rowptr[n] = lo;
}

// ---------------- per-(node,head) softmax max & denom ----------------
__global__ void k_stats(const int* __restrict__ src, const int* __restrict__ rowptr,
                        const float* __restrict__ el, const float* __restrict__ er,
                        float* __restrict__ mmax, float* __restrict__ denom) {
  int idx = blockIdx.x * blockDim.x + threadIdx.x;
  if (idx >= NNODES * NHEAD) return;
  int n = idx >> 2, hh = idx & 3;
  int s = rowptr[n], e = rowptr[n + 1];
  float ern = er[idx];
  float m = -INFINITY;
  for (int i = s; i < e; ++i) {
    float v = el[src[i] * NHEAD + hh] + ern;
    v = v > 0.0f ? v : NEG_SLOPE * v;
    m = fmaxf(m, v);
  }
  float sum = 0.f;
  for (int i = s; i < e; ++i) {
    float v = el[src[i] * NHEAD + hh] + ern;
    v = v > 0.0f ? v : NEG_SLOPE * v;
    sum += __expf(v - m);
  }
  if (e <= s) { m = 0.f; sum = 1.f; }
  mmax[idx] = m;
  denom[idx] = sum;
}

__device__ __forceinline__ void fma4(float4& a, float s, float4 b) {
  a.x = fmaf(s, b.x, a.x); a.y = fmaf(s, b.y, a.y);
  a.z = fmaf(s, b.z, a.z); a.w = fmaf(s, b.w, a.w);
}

// ---------------- aggregation: out[n] = ELU( sum_e alpha*h[src] + bias ) -> bf16 ----------
// One wave per destination node: 32 lanes x 16 floats = 512 accumulators;
// h (41 MB) is L2-resident on the 192 MB L2, gathers are b128.
__global__ __launch_bounds__(256) void k_aggregate(
    const int* __restrict__ src, const int* __restrict__ rowptr,
    const float* __restrict__ h, const float* __restrict__ el,
    const float* __restrict__ er, const float* __restrict__ mmax,
    const float* __restrict__ denom, const float* __restrict__ bias,
    __bf16* __restrict__ outb)
{
  int n = blockIdx.x * (blockDim.x >> 5) + (threadIdx.x >> 5);
  if (n >= NNODES) return;
  int lane = threadIdx.x & 31;
  int head = lane >> 3;                  // dims [lane*16, lane*16+16) -> head lane/8
  int s = rowptr[n], e = rowptr[n + 1];
  int ih = n * NHEAD + head;
  float m = mmax[ih];
  float inv_dn = 1.0f / denom[ih];
  float ern = er[ih];
  const int off = lane << 4;

  float4 acc[4] = {{0,0,0,0},{0,0,0,0},{0,0,0,0},{0,0,0,0}};
  for (int i = s; i < e; ++i) {
    int sv = src[i];
    float v = el[sv * NHEAD + head] + ern;
    v = v > 0.0f ? v : NEG_SLOPE * v;
    float alpha = __expf(v - m) * inv_dn;
    const float4* hp = (const float4*)(h + (size_t)sv * HDDIM + off);
    fma4(acc[0], alpha, hp[0]);
    fma4(acc[1], alpha, hp[1]);
    fma4(acc[2], alpha, hp[2]);
    fma4(acc[3], alpha, hp[3]);
  }

  const float* bp = bias + off;
  __bf16* op = outb + (size_t)n * HDDIM + off;
#pragma unroll
  for (int j = 0; j < 4; ++j) {
    float v0 = acc[j].x + bp[j * 4 + 0];
    float v1 = acc[j].y + bp[j * 4 + 1];
    float v2 = acc[j].z + bp[j * 4 + 2];
    float v3 = acc[j].w + bp[j * 4 + 3];
    v0 = v0 > 0.f ? v0 : (__expf(v0) - 1.f);
    v1 = v1 > 0.f ? v1 : (__expf(v1) - 1.f);
    v2 = v2 > 0.f ? v2 : (__expf(v2) - 1.f);
    v3 = v3 > 0.f ? v3 : (__expf(v3) - 1.f);
    op[j * 4 + 0] = (__bf16)v0;
    op[j * 4 + 1] = (__bf16)v1;
    op[j * 4 + 2] = (__bf16)v2;
    op[j * 4 + 3] = (__bf16)v3;
  }
}

// ---------------- launch ----------------
extern "C" void kernel_launch(void* const* d_in, const int* in_sizes, int n_in,
                              void* d_out, int out_size, void* d_ws, size_t ws_size,
                              hipStream_t stream) {
  (void)in_sizes; (void)n_in; (void)out_size; (void)ws_size;
  const float* feature = (const float*)d_in[0];
  const int*   src     = (const int*)d_in[1];
  const int*   dst     = (const int*)d_in[2];
  const float* W1      = (const float*)d_in[3];
  const float* al1     = (const float*)d_in[4];
  const float* ar1     = (const float*)d_in[5];
  const float* b1      = (const float*)d_in[6];
  const float* W2      = (const float*)d_in[7];
  const float* al2     = (const float*)d_in[8];
  const float* ar2     = (const float*)d_in[9];
  const float* b2      = (const float*)d_in[10];
  const float* Wfc     = (const float*)d_in[11];
  const float* bfc     = (const float*)d_in[12];
  float* out = (float*)d_out;

  char* ws = (char*)d_ws;
  size_t off = 0;
  auto alloc = [&](size_t bytes) -> void* {
    void* p = ws + off;
    off = (off + bytes + 255) & ~(size_t)255;
    return p;
  };
  __bf16* xb    = (__bf16*)alloc((size_t)NNODES * INF_DIM * 2);
  __bf16* wt1   = (__bf16*)alloc((size_t)HDDIM * INF_DIM * 2);
  __bf16* wt2   = (__bf16*)alloc((size_t)HDDIM * HDDIM * 2);
  __bf16* wtfc  = (__bf16*)alloc((size_t)FC_OUT * HDDIM * 2);
  float*  hbuf  = (float*)alloc((size_t)NNODES * HDDIM * 4);
  __bf16* actb  = (__bf16*)alloc((size_t)NNODES * HDDIM * 2);
  float*  el    = (float*)alloc((size_t)NNODES * NHEAD * 4);
  float*  er    = (float*)alloc((size_t)NNODES * NHEAD * 4);
  float*  mm    = (float*)alloc((size_t)NNODES * NHEAD * 4);
  float*  dn    = (float*)alloc((size_t)NNODES * NHEAD * 4);
  int*    rowp  = (int*)alloc((size_t)(NNODES + 1) * 4);

  // one-time conversions
  {
    int c = NNODES * INF_DIM;
    k_cvt_bf16<<<(c + 255) / 256, 256, 0, stream>>>(feature, xb, c);
  }
  k_cvt_transpose_bf16<<<(INF_DIM * HDDIM + 255) / 256, 256, 0, stream>>>(W1, wt1, INF_DIM, HDDIM);
  k_cvt_transpose_bf16<<<(HDDIM * HDDIM + 255) / 256, 256, 0, stream>>>(W2, wt2, HDDIM, HDDIM);
  k_cvt_transpose_bf16<<<(HDDIM * FC_OUT + 255) / 256, 256, 0, stream>>>(Wfc, wtfc, HDDIM, FC_OUT);
  k_rowptr<<<(NNODES + 1 + 255) / 256, 256, 0, stream>>>(dst, rowp);

  const int nh_blocks = (NNODES * NHEAD + 255) / 256;
  const int agg_blocks = (NNODES + 7) / 8;

  // ---- GAT layer 1 ----
  {
    int total = (NNODES / 32) * (HDDIM / 64);   // 5000 waves
    k_gemm_wmma<<<(total + 3) / 4, 128, 0, stream>>>(xb, wt1, nullptr, hbuf,
                                                     NNODES, HDDIM, INF_DIM, 0);
  }
  k_scores<<<nh_blocks, 256, 0, stream>>>(hbuf, al1, ar1, el, er);
  k_stats<<<nh_blocks, 256, 0, stream>>>(src, rowp, el, er, mm, dn);
  k_aggregate<<<agg_blocks, 256, 0, stream>>>(src, rowp, hbuf, el, er, mm, dn, b1, actb);

  // ---- GAT layer 2 ----
  {
    int total = (NNODES / 32) * (HDDIM / 64);
    k_gemm_wmma<<<(total + 3) / 4, 128, 0, stream>>>(actb, wt2, nullptr, hbuf,
                                                     NNODES, HDDIM, HDDIM, 0);
  }
  k_scores<<<nh_blocks, 256, 0, stream>>>(hbuf, al2, ar2, el, er);
  k_stats<<<nh_blocks, 256, 0, stream>>>(src, rowp, el, er, mm, dn);
  k_aggregate<<<agg_blocks, 256, 0, stream>>>(src, rowp, hbuf, el, er, mm, dn, b2, actb);

  // ---- FC head: bias + ELU fused ----
  {
    int total = (NNODES / 32) * (FC_OUT / 64);  // 625 waves
    k_gemm_wmma<<<(total + 3) / 4, 128, 0, stream>>>(actb, wtfc, bfc, out,
                                                     NNODES, FC_OUT, HDDIM, 3);
  }
}
